// DAWN_12979391168723
// MI455X (gfx1250) — matrix operand order repaired and checked by reading, not compiled, not run on hardware
//
#include <hip/hip_runtime.h>
#include <hip/hip_bf16.h>

// ---------------------------------------------------------------------------
// Problem constants (match reference)
// ---------------------------------------------------------------------------
#define BB     8
#define SEQ    2048
#define DM     1024
#define DSP    64
#define NTOT   928      // only first 928 neurons (qk|fv|rel|val) are consumed
#define LSTR   944      // padded f32 logits row stride (elements)

typedef __attribute__((ext_vector_type(16))) __bf16        v16bf;
typedef __attribute__((ext_vector_type(8)))  float         v8f;
typedef __attribute__((ext_vector_type(4)))  unsigned int  v4u;
typedef __attribute__((ext_vector_type(8)))  int           v8i;
typedef __attribute__((ext_vector_type(4)))  int           v4i;

// ---------------------------------------------------------------------------
// bf16 helpers (bit-level RNE)
// ---------------------------------------------------------------------------
__device__ __forceinline__ unsigned short f2bfbits(float f) {
    unsigned int x = __float_as_uint(f);
    return (unsigned short)((x + 0x7FFFu + ((x >> 16) & 1u)) >> 16);
}
__device__ __forceinline__ __bf16 f2bf(float f) {
    union { unsigned short s; __bf16 b; } u;
    u.s = f2bfbits(f);
    return u.b;
}

// ---------------------------------------------------------------------------
// Prep: W (f32 [1024][64]) -> bf16 B-fragment-major layout:
//   Wf[kstep 0..31][ntile 0..3][lane 0..31][e 0..15]
//   value = W[(kstep*32 + (lane<16?0:8) + (e&7) + (e&8?16:0))*64 + ntile*16 + (lane&15)]
// Each WMMA lane then loads its 32B fragment with two b128 loads.
// ---------------------------------------------------------------------------
__global__ void __launch_bounds__(256) dawn_prep_w(const float* __restrict__ W,
                                                   __bf16* __restrict__ Wf) {
    int o  = blockIdx.x * 256 + threadIdx.x;     // 65536 total
    int e  = o & 15;
    int ln = (o >> 4) & 31;
    int nt = (o >> 9) & 3;
    int ks = o >> 11;
    int K  = ks * 32 + ((ln < 16) ? 0 : 8) + (e & 7) + ((e & 8) ? 16 : 0);
    int n  = nt * 16 + (ln & 15);
    Wf[o] = f2bf(W[K * DSP + n]);
}

// ---------------------------------------------------------------------------
// Prep: normalize neuron_emb rows -> bf16 B-fragment-major layout for GEMM2:
//   embF[tile 0..57][kt 0..1][lane 0..31][e 0..15]
// One wave per neuron; lane handles d=lane and d=lane+32.
// ---------------------------------------------------------------------------
__global__ void __launch_bounds__(32) dawn_prep_emb(const float* __restrict__ emb,
                                                    __bf16* __restrict__ embF) {
    int n = blockIdx.x;                 // 0..927
    int lane = threadIdx.x;             // 0..31
    float v0 = emb[n * DSP + lane];
    float v1 = emb[n * DSP + 32 + lane];
    float ss = v0 * v0 + v1 * v1;
    #pragma unroll
    for (int off = 16; off; off >>= 1) ss += __shfl_xor(ss, off);
    float rn = rsqrtf(ss);
    int tile = n >> 4, nl = n & 15;
    int kd = lane;                               // K within 32-wide k-tile
    int fl = nl + ((kd & 8) ? 16 : 0);           // fragment lane
    int e  = (kd & 7) + ((kd & 16) ? 8 : 0);     // fragment element
    embF[((tile * 2 + 0) * 32 + fl) * 16 + e] = f2bf(v0 * rn);  // kt = 0
    embF[((tile * 2 + 1) * 32 + fl) * 16 + e] = f2bf(v1 * rn);  // kt = 1
}

__global__ void __launch_bounds__(256) dawn_zero(float* __restrict__ w_all) {
    int i = blockIdx.x * 256 + threadIdx.x;
    if (i < BB * NTOT) w_all[i] = 0.f;
}

// ---------------------------------------------------------------------------
// Fused main kernel, one block = 16 tokens, 8 waves, dynamic LDS:
//   [0 .. 64K)   xf   : f32 x tile (filled by TDM tensor_load_to_lds)
//                       ... later reused as f32 logits [16][944]
//   [64K .. 96K) xfrag: bf16 A-fragments [kstep 32][lane 32][16]
// ---------------------------------------------------------------------------
__global__ void __launch_bounds__(256) dawn_main(const float*  __restrict__ x,
                                                 const float*  __restrict__ imp,
                                                 const __bf16* __restrict__ Wf,
                                                 const float*  __restrict__ bias,
                                                 const __bf16* __restrict__ embF,
                                                 float*        __restrict__ w_all) {
    extern __shared__ unsigned char dyn[];
    float*  xf    = (float*)dyn;               // 65536 B
    __bf16* xfrag = (__bf16*)(dyn + 65536);    // 32768 B
    float*  lgf   = (float*)dyn;               // alias of xf (time-disjoint)

    __shared__ float  part[4 * 16 * 16];
    __shared__ __bf16 hl[16 * DSP];
    __shared__ float  gmax[16 * 4];
    __shared__ float  ginv[16 * 4];

    const int t     = threadIdx.x;
    const int lane  = t & 31;
    const int wv    = t >> 5;
    const int m0    = blockIdx.x * 16;
    const int batch = m0 >> 11;                // tiles never straddle batches
    const int srow  = m0 & (SEQ - 1);

    // ---- TDM: async DMA the contiguous 64KB x tile into LDS (wave 0) ----
    if (wv == 0) {
        unsigned long long ga = (unsigned long long)(const void*)(x + (size_t)m0 * DM);
        unsigned int lds_off  = (unsigned int)(unsigned long long)(const void*)xf;
        v4u g0;
        g0[0] = 1u;                                     // count=1, user mode
        g0[1] = lds_off;                                // lds_addr
        g0[2] = (unsigned int)ga;                       // global_addr[31:0]
        g0[3] = ((unsigned int)(ga >> 32) & 0x01FFFFFFu) | 0x80000000u; // [56:32]|type=2
        v8i g1;
        g1[0] = 0x00020000;                  // workgroup_mask=0, data_size=2 (4B)
        g1[1] = (int)((16777216u & 0xFFFFu) << 16);  // tensor_dim0[15:0] << 16
        g1[2] = (int)((16777216u >> 16) & 0xFFFFu);  // tensor_dim0[31:16]
        g1[3] = (int)(16384u << 16);                 // tile_dim0 = 16384 elems
        g1[4] = 0;                                   // tile_dim1/2 unused
        g1[5] = (int)16777216u;                      // tensor_dim0_stride lo32
        g1[6] = 0;
        g1[7] = 0;
        v4i gz = {0, 0, 0, 0};
#if __clang_major__ >= 23
        v8i gz8 = {0, 0, 0, 0, 0, 0, 0, 0};
        __builtin_amdgcn_tensor_load_to_lds(g0, g1, gz, gz, gz8, 0);
#else
        __builtin_amdgcn_tensor_load_to_lds(g0, g1, gz, gz, 0);
#endif
        __builtin_amdgcn_s_wait_tensorcnt(0);          // per-wave counter
    }
    // warm L2 for the next block's tile once this tile's DMA is done
    if (t < 16) __builtin_prefetch(x + (size_t)(m0 + 16) * DM + t * 256, 0, 1);
    __syncthreads();

    // ---- LDS f32 -> LDS bf16 A-fragment layout (one b64 store per quad) ----
    for (int q = t; q < 16 * DM / 4; q += 256) {
        int m = q >> 8;                 // row (1024/4 = 256 quads per row)
        int k = (q & 255) * 4;          // column base
        float4 v = *(const float4*)(xf + m * DM + k);
        int kstep = k >> 5, kk = k & 31;
        int fl = m + (((kk >> 3) & 1) << 4);              // fragment lane
        int e  = (kk & 7) + ((kk & 16) ? 8 : 0);          // fragment elem (4-aligned)
        union { unsigned short s[4]; unsigned long long u; } p;
        p.s[0] = f2bfbits(v.x); p.s[1] = f2bfbits(v.y);
        p.s[2] = f2bfbits(v.z); p.s[3] = f2bfbits(v.w);
        *(unsigned long long*)(xfrag + ((kstep * 32 + fl) << 4) + e) = p.u;
    }
    __syncthreads();

    // ---- GEMM1: h = x@W (+b).  wave -> (ntile = wv&3, khalf = wv>>2) ----
    const int ntile = wv & 3;
    const int khalf = wv >> 2;
    const int mrow  = lane & 15;
    const int ncol1 = ntile * 16 + (lane & 15);

    v8f acc = {0.f, 0.f, 0.f, 0.f, 0.f, 0.f, 0.f, 0.f};
    for (int kk = 0; kk < 16; ++kk) {
        const int kstep = khalf * 16 + kk;
        v16bf a  = *(const v16bf*)(xfrag + ((kstep * 32 + lane) << 4));      // 2x ds_load_b128
        v16bf bf = *(const v16bf*)(Wf + (((kstep * 4 + ntile) * 32 + lane) << 4)); // 2x global b128
        acc = __builtin_amdgcn_wmma_f32_16x16x32_bf16(
                  false, a, false, bf, (short)0, acc, false, false);
    }
    if (khalf == 1) {
        #pragma unroll
        for (int i = 0; i < 8; ++i) {
            int M = i + ((lane >> 4) << 3);
            part[(ntile * 16 + M) * 16 + (lane & 15)] = acc[i];
        }
    }
    __syncthreads();
    if (khalf == 0) {
        float bv = bias[ncol1];
        #pragma unroll
        for (int i = 0; i < 8; ++i) {
            int M = i + ((lane >> 4) << 3);
            float h = acc[i] + part[(ntile * 16 + M) * 16 + (lane & 15)] + bv;
            hl[M * DSP + ncol1] = f2bf(h);
        }
    }
    __syncthreads();

    // ---- GEMM2: logits = h @ embT, 58 N-tiles, K=64 (2 WMMAs/tile) ----
    const int kbA = (lane < 16) ? 0 : 8;
    v16bf a0, a1;
    #pragma unroll
    for (int e = 0; e < 8; ++e) {
        a0[e]     = hl[mrow * DSP + kbA + e];
        a0[8 + e] = hl[mrow * DSP + 16 + kbA + e];
        a1[e]     = hl[mrow * DSP + 32 + kbA + e];
        a1[8 + e] = hl[mrow * DSP + 48 + kbA + e];
    }
    for (int tile = wv; tile < NTOT / 16; tile += 8) {
        const int nc = tile * 16 + (lane & 15);
        v16bf b0 = *(const v16bf*)(embF + (((tile * 2 + 0) * 32 + lane) << 4));
        v16bf b1 = *(const v16bf*)(embF + (((tile * 2 + 1) * 32 + lane) << 4));
        v8f c = {0.f, 0.f, 0.f, 0.f, 0.f, 0.f, 0.f, 0.f};
        c = __builtin_amdgcn_wmma_f32_16x16x32_bf16(false, a0, false, b0,
                                                    (short)0, c, false, false);
        c = __builtin_amdgcn_wmma_f32_16x16x32_bf16(false, a1, false, b1,
                                                    (short)0, c, false, false);
        #pragma unroll
        for (int i = 0; i < 8; ++i) {
            int M = i + ((lane >> 4) << 3);
            lgf[M * LSTR + nc] = c[i];              // f32 logits in LDS
        }
    }
    __syncthreads();

    // ---- per-row per-group softmax stats: wave wv owns rows 2wv, 2wv+1 ----
    {
        const int r0 = wv * 2;
        #pragma unroll
        for (int rr = 0; rr < 2; ++rr) {
            const int r = r0 + rr;
            const float ip = imp[(size_t)batch * SEQ + srow + r];
            #pragma unroll
            for (int g = 0; g < 4; ++g) {
                const int lo = (g == 0) ? 0 : (g == 1) ? 512 : (g == 2) ? 768 : 896;
                const int sz = (g == 0) ? 512 : (g == 1) ? 256 : (g == 2) ? 128 : 32;
                float mx = -3.4e38f;
                for (int j = lane; j < sz; j += 32)
                    mx = fmaxf(mx, lgf[r * LSTR + lo + j]);
                #pragma unroll
                for (int off = 16; off; off >>= 1) mx = fmaxf(mx, __shfl_xor(mx, off));
                float sm = 0.f;
                for (int j = lane; j < sz; j += 32)
                    sm += __expf(lgf[r * LSTR + lo + j] - mx);
                #pragma unroll
                for (int off = 16; off; off >>= 1) sm += __shfl_xor(sm, off);
                if (lane == 0) { gmax[r * 4 + g] = mx; ginv[r * 4 + g] = ip / sm; }
            }
        }
    }
    __syncthreads();

    // ---- weighted accumulation over the 16 rows, one atomic per neuron ----
    for (int n = t; n < NTOT; n += 256) {
        const int g = (n < 512) ? 0 : (n < 768) ? 1 : (n < 896) ? 2 : 3;
        float p = 0.f;
        #pragma unroll 4
        for (int r = 0; r < 16; ++r)
            p += ginv[r * 4 + g] * __expf(lgf[r * LSTR + n] - gmax[r * 4 + g]);
        atomicAdd(&w_all[batch * NTOT + n], p);
    }
}

// ---------------------------------------------------------------------------
// Finalize: per-batch top-k / sparsify / sorted-index emission + raw copies.
// Rank(e) = #{j: w[j] > w[e]} + #{j<e: w[j]==w[e]}  (matches jax top_k ties).
// ---------------------------------------------------------------------------
__global__ void __launch_bounds__(256) dawn_finalize(const float* __restrict__ w_all,
                                                     float* __restrict__ out) {
    const int bb = blockIdx.x;          // batch
    const int t  = threadIdx.x;
    __shared__ float wl[NTOT];
    __shared__ int   sel[512];
    const float* wb = w_all + bb * NTOT;
    for (int i = t; i < NTOT; i += 256) wl[i] = wb[i];
    __syncthreads();

    float* o_idx_qk = out;              // [8][64]
    float* o_idx_v  = out + 512;        // [8][32]
    float* o_rwQ    = out + 768;        // [8][128]
    float* o_rwK    = out + 1792;       // [8][128]
    float* o_vw     = out + 2816;       // [8][32]
    float* o_wqk    = out + 3072;       // [8][512]
    float* o_wv     = out + 7168;       // [8][256]

    // --- qk: size 512, K=64, sorted indices + raw copy ---
    for (int e = t; e < 512; e += 256) {
        float v = wl[e]; int rank = 0;
        for (int j = 0; j < 512; ++j) {
            float u = wl[j]; rank += (u > v) || ((u == v) && (j < e));
        }
        sel[e] = (rank < 64);
        o_wqk[bb * 512 + e] = v;
    }
    __syncthreads();
    for (int e = t; e < 512; e += 256) {
        if (sel[e]) {
            int pos = 0;
            for (int j = 0; j < e; ++j) pos += sel[j];
            o_idx_qk[bb * 64 + pos] = (float)e;
        }
    }
    __syncthreads();

    // --- fv: size 256, K=32 ---
    for (int e = t; e < 256; e += 256) {
        float v = wl[512 + e]; int rank = 0;
        for (int j = 0; j < 256; ++j) {
            float u = wl[512 + j]; rank += (u > v) || ((u == v) && (j < e));
        }
        sel[e] = (rank < 32);
        o_wv[bb * 256 + e] = v;
    }
    __syncthreads();
    for (int e = t; e < 256; e += 256) {
        if (sel[e]) {
            int pos = 0;
            for (int j = 0; j < e; ++j) pos += sel[j];
            o_idx_v[bb * 32 + pos] = (float)e;
        }
    }
    __syncthreads();

    // --- rel: size 128, K=16, sparsify (Q and K identical in eval) ---
    for (int e = t; e < 128; e += 256) {
        float v = wl[768 + e]; int rank = 0;
        for (int j = 0; j < 128; ++j) {
            float u = wl[768 + j]; rank += (u > v) || ((u == v) && (j < e));
        }
        float o = (rank < 16) ? v : 0.f;
        o_rwQ[bb * 128 + e] = o;
        o_rwK[bb * 128 + e] = o;
    }
    // --- val: size 32, K=3, sparsify ---
    for (int e = t; e < 32; e += 256) {
        float v = wl[896 + e]; int rank = 0;
        for (int j = 0; j < 32; ++j) {
            float u = wl[896 + j]; rank += (u > v) || ((u == v) && (j < e));
        }
        o_vw[bb * 32 + e] = (rank < 3) ? v : 0.f;
    }
}

// ---------------------------------------------------------------------------
// Launcher
// ---------------------------------------------------------------------------
extern "C" void kernel_launch(void* const* d_in, const int* in_sizes, int n_in,
                              void* d_out, int out_size, void* d_ws, size_t ws_size,
                              hipStream_t stream) {
    (void)in_sizes; (void)n_in; (void)out_size; (void)ws_size;
    const float* x    = (const float*)d_in[0];   // [8,2048,1024]
    const float* imp  = (const float*)d_in[1];   // [8,2048]
    const float* W    = (const float*)d_in[2];   // [1024,64]
    const float* bias = (const float*)d_in[3];   // [64]
    const float* emb  = (const float*)d_in[4];   // [1184,64] (first 928 used)

    // workspace layout
    __bf16* Wf    = (__bf16*)d_ws;                                   // 131072 B
    __bf16* embF  = (__bf16*)((char*)d_ws + 131072);                 // 118784 B
    float*  w_all = (float*)((char*)d_ws + 131072 + 118784);         //  29696 B

    dawn_prep_w   <<<256, 256, 0, stream>>>(W, Wf);
    dawn_prep_emb <<<NTOT, 32, 0, stream>>>(emb, embF);
    dawn_zero     <<<(BB * NTOT + 255) / 256, 256, 0, stream>>>(w_all);
    dawn_main     <<<(BB * SEQ) / 16, 256, 98304, stream>>>(x, imp, Wf, bias, embF, w_all);
    dawn_finalize <<<BB, 256, 0, stream>>>(w_all, (float*)d_out);
}
